// RoIProcessingNetwork_3109556322620
// MI455X (gfx1250) — matrix operand order repaired and checked by reading, not compiled, not run on hardware
//
#include <hip/hip_runtime.h>
#include <hip/hip_bf16.h>

// ---------------- types ----------------
typedef __bf16 bf16_t;
typedef __attribute__((ext_vector_type(16))) __bf16 v16bf;
typedef __attribute__((ext_vector_type(8)))  float  v8f;

// ---------------- problem constants ----------------
#define R_ROIS   2000
#define K_FC1    12544        // 256*7*7
#define CLIN     1024
#define NCLS_PAD 128          // 81 padded to 128 (multiple of 64)
#define NREG     320
#define NC       80
#define NCAND    (R_ROIS * NC)   // 160000
#define SELCAP   2048
#define PRENMS   1000
#define IMTOP    100
#define N_IMG    2
#define LDT      40           // padded LDS row stride (bf16 elems): 80B rows, conflict-free

// =====================================================================
// Tiled bf16 WMMA GEMM:  C[M x N] = act( A[M x K] * B[N x K]^T + bias )
// A row-major (lda = K), B stored N x K row-major (ldb = K) -> K contiguous.
// Block tile 128(M) x BN(N), K step 32, 8 waves as 4(M) x 2(N) groups.
// BN=128: each wave owns 32x64 = 2x4 WMMA tiles (8 wmma / K-step).
// BN=64 : each wave owns 32x32 = 2x2 WMMA tiles (heads).
// Staging uses CDNA5 GLOBAL_LOAD_ASYNC_TO_LDS_B128 (ASYNCcnt), double
// buffered; fragments read back with conflict-free ds_load_b128.
// OUTBF/RELU are compile-time to keep the epilogue spill-free.
// =====================================================================
template <int BN, bool OUTBF, bool RELU>
__global__ void __launch_bounds__(256)
gemm_bf16_wmma(const bf16_t* __restrict__ A, int lda,
               const bf16_t* __restrict__ B, int ldb,
               const float*  __restrict__ bias,
               void* __restrict__ outP, int ldc,
               int M, int N, int K)
{
    constexpr int NFRAG = BN / 32;          // N frags per wave (2 or 4)
    __shared__ __align__(16) bf16_t sA[2][128 * LDT];
    __shared__ __align__(16) bf16_t sB[2][BN * LDT];

    const int tid    = threadIdx.x;
    const int blockM = blockIdx.x * 128;
    const int blockN = blockIdx.y * BN;

    const int lane  = tid & 31;
    const int wave  = tid >> 5;
    const int wm    = wave & 3;    // 0..3 : rows wm*32 .. +32
    const int wn    = wave >> 2;   // 0..1 : cols wn*(BN/2) .. +(BN/2)
    const int lrow  = lane & 15;
    const int lhalf = lane >> 4;

    // staging roles (A: 128x32 tile; thread -> one 32B chunk)
    const int arow  = tid & 127;
    const int ahalf = tid >> 7;           // 0/1 -> k cols 16*ahalf..+15
    int agrow = blockM + arow; if (agrow > M - 1) agrow = M - 1;  // clamp tail

    const v8f vzero = {0.f,0.f,0.f,0.f,0.f,0.f,0.f,0.f};
    v8f acc[2][NFRAG];
    for (int s = 0; s < 2; ++s)
        for (int t2 = 0; t2 < NFRAG; ++t2)
            acc[s][t2] = vzero;

    // Async global -> LDS staging (ASYNCcnt path). The 24-bit instruction
    // offset is applied to BOTH the LDS and the global address (ISA 8.4.4),
    // so one address pair covers each 32B chunk with offset:16.
    auto stage = [&](int buf, int k0) {
        {
            unsigned long long ga =
                (unsigned long long)(A + (size_t)agrow * lda + k0 + 16 * ahalf);
            unsigned la = (unsigned)(size_t)&sA[buf][arow * LDT + 16 * ahalf];
            asm volatile(
                "global_load_async_to_lds_b128 %0, %1, off\n\t"
                "global_load_async_to_lds_b128 %0, %1, off offset:16"
                :: "v"(la), "v"(ga) : "memory");
            __builtin_prefetch(A + (size_t)agrow * lda + k0 + 128, 0, 1);
        }
        if constexpr (BN == 128) {
            const int brow = tid & 127;        // 32B chunk per thread
            const int bhalf = tid >> 7;
            unsigned long long gb =
                (unsigned long long)(B + (size_t)(blockN + brow) * ldb + k0 + 16 * bhalf);
            unsigned lb = (unsigned)(size_t)&sB[buf][brow * LDT + 16 * bhalf];
            asm volatile(
                "global_load_async_to_lds_b128 %0, %1, off\n\t"
                "global_load_async_to_lds_b128 %0, %1, off offset:16"
                :: "v"(lb), "v"(gb) : "memory");
        } else {
            const int brow = tid & 63;         // 16B chunk per thread
            const int bseg = tid >> 6;         // 0..3 -> k cols 8*bseg..+7
            unsigned long long gb =
                (unsigned long long)(B + (size_t)(blockN + brow) * ldb + k0 + 8 * bseg);
            unsigned lb = (unsigned)(size_t)&sB[buf][brow * LDT + 8 * bseg];
            asm volatile(
                "global_load_async_to_lds_b128 %0, %1, off"
                :: "v"(lb), "v"(gb) : "memory");
        }
    };

    const int nk = K >> 5;
    stage(0, 0);
    asm volatile("s_wait_asynccnt 0x0" ::: "memory");
    __syncthreads();
    int buf = 0;

    for (int kt = 0; kt < nk; ++kt) {
        if (kt + 1 < nk) stage(buf ^ 1, (kt + 1) << 5);

        // --- fragments from LDS per documented 16-bit WMMA layouts ---
        // A 16x32: lane m=L%16, half=L/16; VGPR0-3 -> k = 8*half + 0..7,
        //          VGPR4-7 -> k = 16 + 8*half + 0..7
        v16bf afr[2];
        for (int s = 0; s < 2; ++s) {
            const bf16_t* base = &sA[buf][(wm * 32 + s * 16 + lrow) * LDT];
            union { v16bf v; uint4 q[2]; } u;
            u.q[0] = *(const uint4*)(base + 8 * lhalf);
            u.q[1] = *(const uint4*)(base + 16 + 8 * lhalf);
            afr[s] = u.v;
        }
        // B 32x16: lane n=L%16, half=L/16; VGPR0-3 -> k = 8*half + 0..7,
        //          VGPR4-7 -> k = 16 + 8*half + 0..7
        v16bf bfr[NFRAG];
        for (int t2 = 0; t2 < NFRAG; ++t2) {
            const bf16_t* base =
                &sB[buf][(wn * (BN / 2) + t2 * 16 + lrow) * LDT];
            union { v16bf v; uint4 q[2]; } u;
            u.q[0] = *(const uint4*)(base + 8 * lhalf);
            u.q[1] = *(const uint4*)(base + 16 + 8 * lhalf);
            bfr[t2] = u.v;
        }

        for (int s = 0; s < 2; ++s)
            for (int t2 = 0; t2 < NFRAG; ++t2)
                acc[s][t2] = __builtin_amdgcn_wmma_f32_16x16x32_bf16(
                    false, afr[s], false, bfr[t2], (short)0, acc[s][t2],
                    false, false);

        asm volatile("s_wait_asynccnt 0x0" ::: "memory");
        __syncthreads();
        buf ^= 1;
    }

    // --- epilogue: C/D layout: VGPR j -> M = j + 8*half; N = lane%16 ---
    bf16_t* outBf = (bf16_t*)outP;
    float*  outF  = (float*)outP;
    for (int s = 0; s < 2; ++s) {
        for (int t2 = 0; t2 < NFRAG; ++t2) {
            const int n  = blockN + wn * (BN / 2) + t2 * 16 + lrow;
            const float bv = bias[n];
            const int m0 = blockM + wm * 32 + s * 16 + lhalf * 8;
            for (int j = 0; j < 8; ++j) {
                const int m = m0 + j;
                if (m < M) {
                    float v = acc[s][t2][j] + bv;
                    if (RELU) v = v > 0.f ? v : 0.f;
                    if (OUTBF) outBf[(size_t)m * ldc + n] = (bf16_t)v;
                    else       outF [(size_t)m * ldc + n] = v;
                }
            }
        }
    }
}

// =====================================================================
// Elementwise f32 -> bf16 cast (grid-stride)
// =====================================================================
__global__ void __launch_bounds__(256)
cast_f32_bf16(const float* __restrict__ src, bf16_t* __restrict__ dst, int n)
{
    int i = blockIdx.x * blockDim.x + threadIdx.x;
    const int stride = gridDim.x * blockDim.x;
    for (; i < n; i += stride) dst[i] = (bf16_t)src[i];
}

// cls weights: (81 x 1024) -> zero-padded (128 x 1024) bf16
__global__ void __launch_bounds__(256)
pad_cls_w(const float* __restrict__ src, bf16_t* __restrict__ dst)
{
    int i = blockIdx.x * blockDim.x + threadIdx.x;
    if (i >= NCLS_PAD * CLIN) return;
    int row = i >> 10;
    dst[i] = (row < 81) ? (bf16_t)src[i] : (bf16_t)0.0f;
}

__global__ void pad_cls_b(const float* __restrict__ src, float* __restrict__ dst)
{
    int i = threadIdx.x;
    if (i < NCLS_PAD) dst[i] = (i < 81) ? src[i] : 0.0f;
}

// =====================================================================
// Softmax + box decode/clip/mask.  1 block per ROI, 128 threads.
// =====================================================================
__global__ void __launch_bounds__(128)
postprocess(const float* __restrict__ logits,   // [R][128]
            const float* __restrict__ regs,     // [R][320]
            const float* __restrict__ proposals,// [R][4]
            const int*   __restrict__ imidx,    // [R]
            const float* __restrict__ imsizes,  // [2][2] (h, w)
            float* __restrict__ cscore,         // [R*80]
            float* __restrict__ cbox)           // [R*80][4]
{
    const int r = blockIdx.x;
    const int t = threadIdx.x;
    __shared__ float red[128];
    __shared__ float sm, sd;

    float l = (t < 81) ? logits[r * NCLS_PAD + t] : -3.0e38f;
    red[t] = l; __syncthreads();
    for (int o = 64; o > 0; o >>= 1) {
        if (t < o) red[t] = fmaxf(red[t], red[t + o]);
        __syncthreads();
    }
    if (t == 0) sm = red[0];
    __syncthreads();
    float e = (t < 81) ? __expf(l - sm) : 0.0f;
    red[t] = e; __syncthreads();
    for (int o = 64; o > 0; o >>= 1) {
        if (t < o) red[t] += red[t + o];
        __syncthreads();
    }
    if (t == 0) sd = red[0];
    __syncthreads();
    if (t >= NC) return;

    const float score = e / sd;
    const float px1 = proposals[r * 4 + 0], py1 = proposals[r * 4 + 1];
    const float px2 = proposals[r * 4 + 2], py2 = proposals[r * 4 + 3];
    const float pw = px2 - px1, ph = py2 - py1;
    const float pcx = px1 + 0.5f * pw, pcy = py1 + 0.5f * ph;
    const float clipv = 4.135166556742356f;  // log(1000/16)
    const float dx = regs[r * NREG + t * 4 + 0] * 0.1f;
    const float dy = regs[r * NREG + t * 4 + 1] * 0.1f;
    const float dw = fminf(regs[r * NREG + t * 4 + 2] * 0.2f, clipv);
    const float dh = fminf(regs[r * NREG + t * 4 + 3] * 0.2f, clipv);
    const float cx = dx * pw + pcx, cy = dy * ph + pcy;
    const float w = __expf(dw) * pw, h = __expf(dh) * ph;
    float bx1 = cx - 0.5f * w, by1 = cy - 0.5f * h;
    float bx2 = cx + 0.5f * w, by2 = cy + 0.5f * h;
    const int img = imidx[r];
    const float maxx = imsizes[img * 2 + 1], maxy = imsizes[img * 2 + 0];
    bx1 = fminf(fmaxf(bx1, 0.f), maxx);
    bx2 = fminf(fmaxf(bx2, 0.f), maxx);
    by1 = fminf(fmaxf(by1, 0.f), maxy);
    by2 = fminf(fmaxf(by2, 0.f), maxy);
    const float bw = bx2 - bx1, bh = by2 - by1;
    const bool ok = (score > 0.05f) && (bw >= 1.0f) && (bh >= 1.0f);
    const int ci = r * NC + t;
    cscore[ci] = ok ? score : -1.0f;
    cbox[ci * 4 + 0] = bx1; cbox[ci * 4 + 1] = by1;
    cbox[ci * 4 + 2] = bx2; cbox[ci * 4 + 3] = by2;
}

// =====================================================================
// Per-image top-~1000 via 1024-bin score histogram + threshold gather.
// 1 block per image.
// =====================================================================
__global__ void __launch_bounds__(256)
select_topk(const float* __restrict__ cscore, const int* __restrict__ imidx,
            float* __restrict__ sel_s, int* __restrict__ sel_i)
{
    const int img = blockIdx.x;
    const int t = threadIdx.x;
    __shared__ int hist[1024];
    __shared__ int thrBin;
    __shared__ int cnt;
    for (int i = t; i < 1024; i += 256) hist[i] = 0;
    if (t == 0) cnt = 0;
    __syncthreads();
    for (int i = t; i < NCAND; i += 256) {
        if (imidx[i / NC] != img) continue;
        float s = cscore[i];
        if (s <= 0.0f) continue;
        int b = (int)(s * 1024.0f);
        b = b > 1023 ? 1023 : (b < 0 ? 0 : b);
        atomicAdd(&hist[b], 1);
    }
    __syncthreads();
    if (t == 0) {
        int total = 0, tb = 0;
        for (int b = 1023; b >= 0; --b) {
            total += hist[b];
            if (total >= PRENMS) { tb = b; break; }
        }
        thrBin = tb;
    }
    __syncthreads();
    const float thr = (float)thrBin / 1024.0f;
    for (int i = t; i < NCAND; i += 256) {
        if (imidx[i / NC] != img) continue;
        float s = cscore[i];
        if (s <= 0.0f || s < thr) continue;
        int slot = atomicAdd(&cnt, 1);
        if (slot < SELCAP) {
            sel_s[img * SELCAP + slot] = s;
            sel_i[img * SELCAP + slot] = i;
        }
    }
    __syncthreads();
    int c = cnt > SELCAP ? SELCAP : cnt;
    for (int i = c + t; i < SELCAP; i += 256) {
        sel_s[img * SELCAP + i] = -1.0e30f;
        sel_i[img * SELCAP + i] = 0;
    }
}

// =====================================================================
// Bitonic sort of 2048 (score desc) in LDS. 1 block / image, 1024 thr.
// =====================================================================
__global__ void __launch_bounds__(1024)
sort_topk(float* __restrict__ sel_s, int* __restrict__ sel_i)
{
    const int img = blockIdx.x;
    const int t = threadIdx.x;
    __shared__ float ss[SELCAP];
    __shared__ int   si[SELCAP];
    for (int i = t; i < SELCAP; i += 1024) {
        ss[i] = sel_s[img * SELCAP + i];
        si[i] = sel_i[img * SELCAP + i];
    }
    __syncthreads();
    for (int k = 2; k <= SELCAP; k <<= 1) {
        for (int j = k >> 1; j > 0; j >>= 1) {
            for (int i = t; i < SELCAP; i += 1024) {
                int ixj = i ^ j;
                if (ixj > i) {
                    bool up = ((i & k) == 0);      // up segment => descending
                    float a = ss[i], b = ss[ixj];
                    bool sw = up ? (a < b) : (a > b);
                    if (sw) {
                        ss[i] = b; ss[ixj] = a;
                        int ti = si[i]; si[i] = si[ixj]; si[ixj] = ti;
                    }
                }
            }
            __syncthreads();
        }
    }
    for (int i = t; i < SELCAP; i += 1024) {
        sel_s[img * SELCAP + i] = ss[i];
        sel_i[img * SELCAP + i] = si[i];
    }
}

// =====================================================================
// Greedy class-aware NMS over top-1000 + emit top-100 (b, s, c).
// 1 block / image, 1024 threads.
// =====================================================================
__global__ void __launch_bounds__(1024)
nms_out(const float* __restrict__ sel_s, const int* __restrict__ sel_i,
        const float* __restrict__ cbox, float* __restrict__ out)
{
    const int img = blockIdx.x;
    const int t = threadIdx.x;
    __shared__ float bx1[PRENMS], by1[PRENMS], bx2[PRENMS], by2[PRENMS];
    __shared__ float ar[PRENMS], sc[PRENMS];
    __shared__ int sup[PRENMS], keep[PRENMS];

    if (t < PRENMS) {
        float s = sel_s[img * SELCAP + t];
        int idx = sel_i[img * SELCAP + t];
        int cls = idx % NC;
        float off = (float)cls * 2048.0f;   // class separation (boxes <= 800)
        float x1 = cbox[idx * 4 + 0] + off;
        float y1 = cbox[idx * 4 + 1] + off;
        float x2 = cbox[idx * 4 + 2] + off;
        float y2 = cbox[idx * 4 + 3] + off;
        bx1[t] = x1; by1[t] = y1; bx2[t] = x2; by2[t] = y2;
        ar[t] = (x2 - x1) * (y2 - y1);
        sc[t] = s;
        sup[t] = (s > 0.0f) ? 0 : 1;
        keep[t] = 0;
    }
    __syncthreads();

    for (int i = 0; i < PRENMS; ++i) {
        bool live = (sup[i] == 0);
        if (live) {
            if (t == 0) keep[i] = 1;
            if (t < PRENMS && t > i) {
                float iw = fminf(bx2[i], bx2[t]) - fmaxf(bx1[i], bx1[t]);
                float ih = fminf(by2[i], by2[t]) - fmaxf(by1[i], by1[t]);
                iw = iw > 0.f ? iw : 0.f;
                ih = ih > 0.f ? ih : 0.f;
                float inter = iw * ih;
                float iou = inter / (ar[i] + ar[t] - inter + 1e-9f);
                if (iou > 0.5f) sup[t] = 1;
            }
        }
        __syncthreads();
    }

    if (t == 0) {
        float* outB = out + img * IMTOP * 4;                // [2][100][4]
        float* outS = out + N_IMG * IMTOP * 4 + img * IMTOP;// [2][100]
        float* outC = out + N_IMG * IMTOP * 5 + img * IMTOP;// [2][100]
        int cnt = 0;
        for (int i = 0; i < PRENMS && cnt < IMTOP; ++i) {
            if (keep[i]) {
                int idx = sel_i[img * SELCAP + i];
                outB[cnt * 4 + 0] = cbox[idx * 4 + 0];
                outB[cnt * 4 + 1] = cbox[idx * 4 + 1];
                outB[cnt * 4 + 2] = cbox[idx * 4 + 2];
                outB[cnt * 4 + 3] = cbox[idx * 4 + 3];
                outS[cnt] = sc[i];
                outC[cnt] = (float)(idx % NC);
                ++cnt;
            }
        }
        for (; cnt < IMTOP; ++cnt) {
            outB[cnt * 4 + 0] = 0.f; outB[cnt * 4 + 1] = 0.f;
            outB[cnt * 4 + 2] = 0.f; outB[cnt * 4 + 3] = 0.f;
            outS[cnt] = 0.f;
            outC[cnt] = -1.0f;
        }
    }
}

// =====================================================================
// Host-side orchestration
// =====================================================================
extern "C" void kernel_launch(void* const* d_in, const int* in_sizes, int n_in,
                              void* d_out, int out_size, void* d_ws, size_t ws_size,
                              hipStream_t stream)
{
    const float* roi_maps  = (const float*)d_in[0];
    const float* proposals = (const float*)d_in[1];
    const int*   imidx     = (const int*)  d_in[2];
    const float* imsizes   = (const float*)d_in[3];
    const float* fc1_w     = (const float*)d_in[4];
    const float* fc1_b     = (const float*)d_in[5];
    const float* fc2_w     = (const float*)d_in[6];
    const float* fc2_b     = (const float*)d_in[7];
    const float* cls_w     = (const float*)d_in[8];
    const float* cls_b     = (const float*)d_in[9];
    const float* reg_w     = (const float*)d_in[10];
    const float* reg_b     = (const float*)d_in[11];
    float* out = (float*)d_out;

    char* ws = (char*)d_ws;
    size_t off = 0;
    auto alloc = [&](size_t bytes) -> char* {
        char* p = ws + off;
        off += bytes;
        off = (off + 255) & ~(size_t)255;
        return p;
    };

    bf16_t* Xbf    = (bf16_t*)alloc((size_t)R_ROIS * K_FC1 * 2);
    bf16_t* W1bf   = (bf16_t*)alloc((size_t)CLIN * K_FC1 * 2);
    bf16_t* W2bf   = (bf16_t*)alloc((size_t)CLIN * CLIN * 2);
    bf16_t* Wcbf   = (bf16_t*)alloc((size_t)NCLS_PAD * CLIN * 2);
    bf16_t* Wrbf   = (bf16_t*)alloc((size_t)NREG * CLIN * 2);
    float*  clsbp  = (float*) alloc(NCLS_PAD * 4);
    bf16_t* H1bf   = (bf16_t*)alloc((size_t)R_ROIS * CLIN * 2);
    bf16_t* H2bf   = (bf16_t*)alloc((size_t)R_ROIS * CLIN * 2);
    float*  logits = (float*) alloc((size_t)R_ROIS * NCLS_PAD * 4);
    float*  regs   = (float*) alloc((size_t)R_ROIS * NREG * 4);
    float*  cscore = (float*) alloc((size_t)NCAND * 4);
    float*  cbox   = (float*) alloc((size_t)NCAND * 4 * 4);
    float*  sel_s  = (float*) alloc((size_t)N_IMG * SELCAP * 4);
    int*    sel_i  = (int*)   alloc((size_t)N_IMG * SELCAP * 4);

    // 1) casts / padding
    cast_f32_bf16<<<4096, 256, 0, stream>>>(roi_maps, Xbf, R_ROIS * K_FC1);
    cast_f32_bf16<<<4096, 256, 0, stream>>>(fc1_w,    W1bf, CLIN * K_FC1);
    cast_f32_bf16<<<2048, 256, 0, stream>>>(fc2_w,    W2bf, CLIN * CLIN);
    cast_f32_bf16<<<1024, 256, 0, stream>>>(reg_w,    Wrbf, NREG * CLIN);
    pad_cls_w<<<(NCLS_PAD * CLIN + 255) / 256, 256, 0, stream>>>(cls_w, Wcbf);
    pad_cls_b<<<1, 128, 0, stream>>>(cls_b, clsbp);

    // 2) GEMMs (WMMA bf16, f32 accumulate)
    dim3 blk(256);
    gemm_bf16_wmma<128, true, true><<<dim3(16, CLIN / 128), blk, 0, stream>>>(
        Xbf, K_FC1, W1bf, K_FC1, fc1_b, H1bf, CLIN, R_ROIS, CLIN, K_FC1);
    gemm_bf16_wmma<128, true, true><<<dim3(16, CLIN / 128), blk, 0, stream>>>(
        H1bf, CLIN, W2bf, CLIN, fc2_b, H2bf, CLIN, R_ROIS, CLIN, CLIN);
    gemm_bf16_wmma<64, false, false><<<dim3(16, NCLS_PAD / 64), blk, 0, stream>>>(
        H2bf, CLIN, Wcbf, CLIN, clsbp, logits, NCLS_PAD, R_ROIS, NCLS_PAD, CLIN);
    gemm_bf16_wmma<64, false, false><<<dim3(16, NREG / 64), blk, 0, stream>>>(
        H2bf, CLIN, Wrbf, CLIN, reg_b, regs, NREG, R_ROIS, NREG, CLIN);

    // 3) softmax + box decode
    postprocess<<<R_ROIS, 128, 0, stream>>>(logits, regs, proposals, imidx,
                                            imsizes, cscore, cbox);

    // 4) per-image top-k, sort, NMS, emit
    select_topk<<<N_IMG, 256, 0, stream>>>(cscore, imidx, sel_s, sel_i);
    sort_topk<<<N_IMG, 1024, 0, stream>>>(sel_s, sel_i);
    nms_out<<<N_IMG, 1024, 0, stream>>>(sel_s, sel_i, cbox, out);
}
